// QGFDMultiheadAttention_19731079758447
// MI455X (gfx1250) — compile-verified, matrix operand order checked by reading, not run on hardware
//
#include <hip/hip_runtime.h>

#define B_  2
#define L_  2048
#define D_  1024
#define H_  16
#define HD_ 64

typedef __attribute__((ext_vector_type(16))) _Float16     v16h;
typedef __attribute__((ext_vector_type(8)))  _Float16     v8h;
typedef __attribute__((ext_vector_type(4)))  _Float16     v4h;
typedef __attribute__((ext_vector_type(8)))  float        v8f;
typedef __attribute__((ext_vector_type(4)))  unsigned int u32x4;
typedef __attribute__((ext_vector_type(8)))  int          i32x8;
typedef __attribute__((ext_vector_type(4)))  int          i32x4;

#if __has_builtin(__builtin_amdgcn_tensor_load_to_lds)
#define USE_TDM 1
#else
#define USE_TDM 0
#endif

// ---------------------------------------------------------------------------
// fp32 -> f16 conversion (vectorized x4)
// ---------------------------------------------------------------------------
__global__ __launch_bounds__(256) void cvt_f32_f16(const float* __restrict__ in,
                                                   _Float16* __restrict__ out,
                                                   int n4) {
  int i = blockIdx.x * 256 + threadIdx.x;
  if (i < n4) {
    float4 f = ((const float4*)in)[i];
    v4h o;
    o[0] = (_Float16)f.x; o[1] = (_Float16)f.y;
    o[2] = (_Float16)f.z; o[3] = (_Float16)f.w;
    ((v4h*)out)[i] = o;
  }
}

#if USE_TDM
// ---------------------------------------------------------------------------
// Tensor Data Mover: DMA a 2D tile (rows x 32 f16, row stride ld_elems) from
// global memory into contiguous LDS at lds_off.  D# built per CDNA5 ISA ch.8:
//  group0: count=1 | lds_addr | global_addr[56:0] | type=2
//  group1: data_size=1(2B), tile_dim0=32, tile_dim1=rows, dim0_stride=ld,
//          tensor dims huge (tiles never clip), no pad/iterate/multicast.
// Issued by one wave per workgroup; tracked with TENSORcnt.
// 6-arg builtin form (clang-23 / therock-10.0 headers).
// ---------------------------------------------------------------------------
__device__ inline void tdm_load_tile(unsigned lds_off, const _Float16* gptr,
                                     int rows, int ld_elems) {
  union { unsigned u[4]; u32x4 v; } g0;
  union { int i[8]; i32x8 v; } g1;
  const unsigned long long ga = (unsigned long long)(uintptr_t)gptr;
  g0.u[0] = 1u;                                   // count = 1 (valid)
  g0.u[1] = lds_off;                              // LDS byte address
  g0.u[2] = (unsigned)(ga & 0xFFFFFFFFu);         // global_addr[31:0]
  g0.u[3] = (unsigned)((ga >> 32) & 0x1FFFFFFu)   // global_addr[56:32]
          | 0x80000000u;                          // type = 2 ("image")
  const unsigned td0 = 0x7FFFFFFFu, td1 = 0x7FFFFFFFu;  // no OOB clipping
  const unsigned long long s0 = (unsigned long long)ld_elems;
  g1.i[0] = (int)(1u << 16);                      // data_size = 1 -> 2 bytes
  g1.i[1] = (int)((td0 & 0xFFFFu) << 16);         // tensor_dim0[15:0]
  g1.i[2] = (int)((td0 >> 16) | ((td1 & 0xFFFFu) << 16));
  g1.i[3] = (int)((td1 >> 16) | (32u << 16));     // tile_dim0 = 32
  g1.i[4] = rows;                                 // tile_dim1 = rows, tile_dim2 = 0
  g1.i[5] = (int)(s0 & 0xFFFFFFFFu);              // dim0_stride[31:0]
  g1.i[6] = (int)((s0 >> 32) & 0xFFFFu);          // dim0_stride[47:32], dim1_stride=0
  g1.i[7] = 0;
  const i32x4 z4 = {};
  const i32x8 z8 = {};
  __builtin_amdgcn_tensor_load_to_lds(g0.v, g1.v, z4, z4, z8, 0);
}
#endif

// ---------------------------------------------------------------------------
// Tiled WMMA GEMM:  C[M,N] = A[M,K] * B[N,K]^T   (both operands row-major [.,K])
// 256 threads = 8 waves; wave tile 64x32 (4x2 wmma 16x16x32 f16 tiles).
// K-step 32, double-buffered LDS tiles filled by the Tensor Data Mover.
// MODE 0: C f16, normal layout, ldc
// MODE 1: C f16, per-batch transpose  C[b*D + n][l]   (v projection -> vT)
// MODE 2: C f16 = result*scale, per-head score matrix at Cbase + zLocal*L*L
// MODE 3: C f16, transposed per-head  C[z*HD*L + n*L + m]  (P @ v -> vT form)
// MODE 4: C f32 = result + bias[n], normal layout (final output projection)
// ---------------------------------------------------------------------------
template<int TM, int TN, int MODE>
__global__ __launch_bounds__(256) void gemm_kernel(
    const _Float16* __restrict__ Abase, const _Float16* __restrict__ Bbase,
    void* __restrict__ Cbase, int M, int N, int K, int lda, int ldb, int ldc,
    int bh_base, float scale, const float* __restrict__ bias)
{
  constexpr int NWN = TN / 32;       // waves along N
  constexpr int NWM = 8 / NWN;       // waves along M
  static_assert(TM == NWM * 64, "tile mismatch");

  __shared__ _Float16 lA[2][TM * 32];
  __shared__ _Float16 lB[2][TN * 32];

  const int tid  = threadIdx.x;
  const int wid  = tid >> 5;
  const int lane = tid & 31;
  const int wm   = wid / NWN;
  const int wn   = wid % NWN;

  const int z = (int)blockIdx.z + bh_base;  // absolute (b*H + h) for modes 2,3
  const _Float16* A  = Abase;
  const _Float16* Bt = Bbase;
  if constexpr (MODE == 2) {
    const int b = z / H_, h = z % H_;
    A  += (size_t)b * L_ * D_ + (size_t)h * HD_;   // q rows of this head
    Bt += (size_t)b * L_ * D_ + (size_t)h * HD_;   // k rows of this head
  } else if constexpr (MODE == 3) {
    A  += (size_t)blockIdx.z * L_ * L_;            // P, chunk-local slot
    Bt += (size_t)z * HD_ * L_;                    // vT slice [HD, L]
  }

  const int row0 = blockIdx.y * TM;
  const int col0 = blockIdx.x * TN;
  const _Float16* Atile = A  + (size_t)row0 * lda;
  const _Float16* Btile = Bt + (size_t)col0 * ldb;

  v8f acc[4][2];
  for (int i = 0; i < 4; i++)
    for (int j = 0; j < 2; j++)
      acc[i][j] = {};

  const int lrow = lane & 15;
  const int koff = (lane >> 4) << 3;
  const int S = K >> 5;   // number of 32-wide K steps

#if USE_TDM
  if (wid == 0)  // prologue: DMA first K-slab into buffer 0
  {
    tdm_load_tile((unsigned)(uintptr_t)(void*)&lA[0][0], Atile, TM, lda);
    tdm_load_tile((unsigned)(uintptr_t)(void*)&lB[0][0], Btile, TN, ldb);
  }
#endif

  for (int s = 0; s < S; s++) {
    const int cur = s & 1;
#if USE_TDM
    if (wid == 0) __builtin_amdgcn_s_wait_tensorcnt((short)0);
    __syncthreads();   // buffer `cur` ready for all; prior ds reads all retired
    if (wid == 0 && s + 1 < S) {  // overlap next DMA with this slab's WMMAs
      tdm_load_tile((unsigned)(uintptr_t)(void*)&lA[cur ^ 1][0],
                    Atile + (size_t)(s + 1) * 32, TM, lda);
      tdm_load_tile((unsigned)(uintptr_t)(void*)&lB[cur ^ 1][0],
                    Btile + (size_t)(s + 1) * 32, TN, ldb);
    }
#else
    __syncthreads();
    {
      const int k0 = s * 32;
      v8h ra[TM * 4 / 256], rb[(TN * 4 + 255) / 256];
      #pragma unroll
      for (int it = 0; it < TM * 4 / 256; it++) {
        int idx = tid + it * 256, r = idx >> 2, c = (idx & 3) << 3;
        ra[it] = *(const v8h*)&Atile[(size_t)r * lda + k0 + c];
      }
      #pragma unroll
      for (int it = 0; it < (TN * 4 + 255) / 256; it++) {
        int idx = tid + it * 256;
        if (idx < TN * 4) {
          int r = idx >> 2, c = (idx & 3) << 3;
          rb[it] = *(const v8h*)&Btile[(size_t)r * ldb + k0 + c];
        }
      }
      #pragma unroll
      for (int it = 0; it < TM * 4 / 256; it++) {
        int idx = tid + it * 256, r = idx >> 2, c = (idx & 3) << 3;
        *(v8h*)&lA[cur][r * 32 + c] = ra[it];
      }
      #pragma unroll
      for (int it = 0; it < (TN * 4 + 255) / 256; it++) {
        int idx = tid + it * 256;
        if (idx < TN * 4) {
          int r = idx >> 2, c = (idx & 3) << 3;
          *(v8h*)&lB[cur][r * 32 + c] = rb[it];
        }
      }
    }
    __syncthreads();
#endif

    // per-lane fragment gather per the CDNA5 16-bit A-matrix 16x32 layout:
    // lanes 0-15 hold K 0..7 / 16..23, lanes 16-31 hold K 8..15 / 24..31
    v16h af[4], bf[2];
    for (int i = 0; i < 4; i++) {
      const _Float16* p = &lA[cur][(wm * 64 + i * 16 + lrow) * 32 + koff];
      v8h c0 = *(const v8h*)p;
      v8h c1 = *(const v8h*)(p + 16);
      for (int e = 0; e < 8; e++) { af[i][e] = c0[e]; af[i][e + 8] = c1[e]; }
    }
    for (int j = 0; j < 2; j++) {
      const _Float16* p = &lB[cur][(wn * 32 + j * 16 + lrow) * 32 + koff];
      v8h c0 = *(const v8h*)p;
      v8h c1 = *(const v8h*)(p + 16);
      for (int e = 0; e < 8; e++) { bf[j][e] = c0[e]; bf[j][e + 8] = c1[e]; }
    }
    for (int i = 0; i < 4; i++)
      for (int j = 0; j < 2; j++)
        acc[i][j] = __builtin_amdgcn_wmma_f32_16x16x32_f16(
            false, af[i], false, bf[j], (short)0, acc[i][j], false, false);
  }

  // D layout: lane holds n = lane&15, rows m = 8*(lane>>4) + r, r=0..7
  const int dn  = lane & 15;
  const int dmb = (lane >> 4) << 3;
  for (int i = 0; i < 4; i++)
    for (int j = 0; j < 2; j++) {
      const int mb = row0 + wm * 64 + i * 16 + dmb;
      const int nb = col0 + wn * 32 + j * 16 + dn;
      for (int r = 0; r < 8; r++) {
        const float v = acc[i][j][r];
        const int m = mb + r, n = nb;
        if constexpr (MODE == 0) {
          ((_Float16*)Cbase)[(size_t)m * ldc + n] = (_Float16)v;
        } else if constexpr (MODE == 1) {
          const int b = m >> 11, l = m & (L_ - 1);
          ((_Float16*)Cbase)[((size_t)b * D_ + n) * L_ + l] = (_Float16)v;
        } else if constexpr (MODE == 2) {
          _Float16* C = (_Float16*)Cbase + (size_t)blockIdx.z * L_ * L_;
          C[(size_t)m * L_ + n] = (_Float16)(v * scale);
        } else if constexpr (MODE == 3) {
          _Float16* C = (_Float16*)Cbase + (size_t)z * HD_ * L_;
          C[(size_t)n * L_ + m] = (_Float16)v;
        } else {
          ((float*)Cbase)[(size_t)m * ldc + n] = v + bias[n];
        }
      }
    }
}

// ---------------------------------------------------------------------------
// In-place row softmax over f16 rows of length L_: one 256-thread block / row
// ---------------------------------------------------------------------------
__global__ __launch_bounds__(256) void softmax_kernel(_Float16* __restrict__ P) {
  __shared__ float red[256];
  _Float16* p = P + (size_t)blockIdx.x * L_;
  const int t = threadIdx.x;
  float x[L_ / 256];
  float mx = -1e30f;
  for (int i = 0; i < L_ / 256; i++) {
    x[i] = (float)p[t + i * 256];
    mx = fmaxf(mx, x[i]);
  }
  red[t] = mx; __syncthreads();
  for (int s = 128; s > 0; s >>= 1) { if (t < s) red[t] = fmaxf(red[t], red[t + s]); __syncthreads(); }
  mx = red[0]; __syncthreads();
  float sum = 0.f;
  for (int i = 0; i < L_ / 256; i++) { x[i] = __expf(x[i] - mx); sum += x[i]; }
  red[t] = sum; __syncthreads();
  for (int s = 128; s > 0; s >>= 1) { if (t < s) red[t] += red[t + s]; __syncthreads(); }
  const float inv = 1.0f / red[0];
  for (int i = 0; i < L_ / 256; i++) p[t + i * 256] = (_Float16)(x[i] * inv);
}

// ---------------------------------------------------------------------------
// attn_h[b, l, h*HD+d] = 0.5*(v1T + v3T) gathered from transposed [B][D][L]
// ---------------------------------------------------------------------------
__global__ __launch_bounds__(256) void combine_kernel(const _Float16* __restrict__ v1T,
                                                      const _Float16* __restrict__ v3T,
                                                      _Float16* __restrict__ out) {
  size_t i = (size_t)blockIdx.x * 256 + threadIdx.x;   // i = (b*L + l)*D + c
  const int c = (int)(i % D_);
  const size_t bl = i / D_;
  const int l = (int)(bl % L_);
  const int b = (int)(bl / L_);
  const size_t tix = ((size_t)b * D_ + c) * L_ + l;
  out[i] = (_Float16)(0.5f * ((float)v1T[tix] + (float)v3T[tix]));
}

// ---------------------------------------------------------------------------
extern "C" void kernel_launch(void* const* d_in, const int* in_sizes, int n_in,
                              void* d_out, int out_size, void* d_ws, size_t ws_size,
                              hipStream_t stream) {
  (void)in_sizes; (void)n_in; (void)out_size;
  const float* x_q  = (const float*)d_in[0];
  const float* x_kv = (const float*)d_in[1];
  const float* Wq   = (const float*)d_in[2];
  const float* Wk   = (const float*)d_in[3];
  const float* Wv   = (const float*)d_in[4];
  const float* Wo   = (const float*)d_in[5];
  const float* bo   = (const float*)d_in[6];
  float* out = (float*)d_out;

  const size_t BLD = (size_t)B_ * L_ * D_;   // 4M elements
  const size_t DD  = (size_t)D_ * D_;        // 1M elements

  char* base = (char*)d_ws;
  size_t off = 0;
  auto alloc_h = [&](size_t elems) -> _Float16* {
    _Float16* p = (_Float16*)(base + off);
    off += (elems * sizeof(_Float16) + 255) & ~(size_t)255;
    return p;
  };
  _Float16* xq_h  = alloc_h(BLD);
  _Float16* xkv_h = alloc_h(BLD);
  _Float16* Wq_h  = alloc_h(DD);
  _Float16* Wk_h  = alloc_h(DD);
  _Float16* Wv_h  = alloc_h(DD);
  _Float16* Wo_h  = alloc_h(DD);
  _Float16* q_h   = alloc_h(BLD);
  _Float16* k_h   = alloc_h(BLD);
  _Float16* vT_h  = alloc_h(BLD);   // [B][D][L] transposed values
  _Float16* v1T   = alloc_h(BLD);
  _Float16* v2T   = alloc_h(BLD);
  _Float16* v3T   = alloc_h(BLD);
  _Float16* at_h  = alloc_h(BLD);

  // P buffer: as many heads as the remaining workspace fits (8 MB per head)
  const size_t perHead = (size_t)L_ * L_ * sizeof(_Float16);
  size_t rem = (ws_size > off) ? (ws_size - off) : 0;
  int max_heads = (int)(rem / perHead);
  if (max_heads < 1) max_heads = 1;
  if (max_heads > B_ * H_) max_heads = B_ * H_;
  _Float16* P = (_Float16*)(base + off);

  // fp32 -> f16 staging
  cvt_f32_f16<<<dim3((unsigned)(BLD / 1024)), dim3(256), 0, stream>>>(x_q,  xq_h,  (int)(BLD / 4));
  cvt_f32_f16<<<dim3((unsigned)(BLD / 1024)), dim3(256), 0, stream>>>(x_kv, xkv_h, (int)(BLD / 4));
  cvt_f32_f16<<<dim3((unsigned)(DD  / 1024)), dim3(256), 0, stream>>>(Wq, Wq_h, (int)(DD / 4));
  cvt_f32_f16<<<dim3((unsigned)(DD  / 1024)), dim3(256), 0, stream>>>(Wk, Wk_h, (int)(DD / 4));
  cvt_f32_f16<<<dim3((unsigned)(DD  / 1024)), dim3(256), 0, stream>>>(Wv, Wv_h, (int)(DD / 4));
  cvt_f32_f16<<<dim3((unsigned)(DD  / 1024)), dim3(256), 0, stream>>>(Wo, Wo_h, (int)(DD / 4));

  // projections: q, k normal; v stored transposed per batch ([B][D][L])
  const dim3 gproj(D_ / 128, (B_ * L_) / 128, 1);
  gemm_kernel<128, 128, 0><<<gproj, 256, 0, stream>>>(
      xq_h, Wq_h, q_h, B_ * L_, D_, D_, D_, D_, D_, 0, 1.0f, nullptr);
  gemm_kernel<128, 128, 0><<<gproj, 256, 0, stream>>>(
      xkv_h, Wk_h, k_h, B_ * L_, D_, D_, D_, D_, D_, 0, 1.0f, nullptr);
  gemm_kernel<128, 128, 1><<<gproj, 256, 0, stream>>>(
      xkv_h, Wv_h, vT_h, B_ * L_, D_, D_, D_, D_, D_, 0, 1.0f, nullptr);

  // per-head: P = softmax(q k^T * scale); then v1=Pv, v2=Pv1, v3=Pv2
  // (associativity: (0.5 P^3 + 0.5 P) v == 0.5 P(P(P v)) + 0.5 P v)
  const float scale = 0.125f;  // HD^-0.5
  int done = 0;
  while (done < B_ * H_) {
    int nz = B_ * H_ - done;
    if (nz > max_heads) nz = max_heads;

    gemm_kernel<128, 128, 2><<<dim3(L_ / 128, L_ / 128, nz), 256, 0, stream>>>(
        q_h, k_h, P, L_, L_, HD_, D_, D_, L_, done, scale, nullptr);
    softmax_kernel<<<dim3((unsigned)(nz * L_)), 256, 0, stream>>>(P);

    gemm_kernel<256, 64, 3><<<dim3(1, L_ / 256, nz), 256, 0, stream>>>(
        P, vT_h, v1T, L_, HD_, L_, L_, L_, L_, done, 1.0f, nullptr);
    gemm_kernel<256, 64, 3><<<dim3(1, L_ / 256, nz), 256, 0, stream>>>(
        P, v1T, v2T, L_, HD_, L_, L_, L_, L_, done, 1.0f, nullptr);
    gemm_kernel<256, 64, 3><<<dim3(1, L_ / 256, nz), 256, 0, stream>>>(
        P, v2T, v3T, L_, HD_, L_, L_, L_, L_, done, 1.0f, nullptr);

    done += nz;
  }

  // attn output = 0.5*v3 + 0.5*v1, back to [B,L,D] f16
  combine_kernel<<<dim3((unsigned)(BLD / 256)), 256, 0, stream>>>(v1T, v3T, at_h);

  // final projection with bias -> fp32 d_out
  gemm_kernel<128, 128, 4><<<gproj, 256, 0, stream>>>(
      at_h, Wo_h, out, B_ * L_, D_, D_, D_, D_, D_, 0, 1.0f, bo);
}
